// AttnContext_18537078850197
// MI455X (gfx1250) — matrix operand order repaired and checked by reading, not compiled, not run on hardware
//
#include <hip/hip_runtime.h>
#include <hip/hip_bf16.h>
#include <math.h>

typedef __attribute__((ext_vector_type(16))) __bf16 v16bf;
typedef __attribute__((ext_vector_type(8)))  float  v8f;

constexpr int B = 32;
constexpr int S = 8192;
constexpr int D = 512;
constexpr int CHUNKS = 32;            // sequence chunks per batch
constexpr int ROWS   = S / CHUNKS;    // 256 rows per block
constexpr int PART_STRIDE = D + 2;    // [m, l, acc[512]] per partial

// ---------------------------------------------------------------------------
// Kernel 1: per-(batch, chunk) partial flash-softmax-attention.
//   scores via v_wmma_f32_16x16x32_bf16 (B operand = broadcast hidden slice),
//   local softmax stats, local weighted accumulation over D (float2 coalesced).
// ---------------------------------------------------------------------------
__global__ __launch_bounds__(256)
void attn_partial_kernel(const float* __restrict__ hidden,
                         const float* __restrict__ src,
                         float* __restrict__ part) {
    __shared__ __align__(32) __bf16 hidbf[D]; // hidden[b] as bf16
    __shared__ float sc[ROWS];                // raw scores for this chunk
    __shared__ float wbuf[ROWS];              // exp(score - m_local)
    __shared__ float red[256];                // reduction scratch

    const int tid   = threadIdx.x;
    const int blk   = blockIdx.x;
    const int b     = blk / CHUNKS;
    const int chunk = blk % CHUNKS;
    const int row0  = chunk * ROWS;

    const float* __restrict__ hb = hidden + (size_t)b * D;
    const float* __restrict__ sb = src + (size_t)b * S * D;

    // Stage hidden[b] into LDS as bf16 (layout matches B-fragment gathers).
    hidbf[tid]       = (__bf16)hb[tid];
    hidbf[tid + 256] = (__bf16)hb[tid + 256];
    __syncthreads();

    const int wv      = tid >> 5;          // wave id 0..7
    const int lane    = tid & 31;
    const int mrow    = lane & 15;         // A-matrix row within 16-row tile
    const int khalf8  = (lane >> 4) << 3;  // unused: A fragment also uses khalf16
    const int khalf16 = (lane >> 4) << 4;  // B fragment: lanes 16-31 hold K 16..31

    // Each wave: 2 tiles of 16 rows -> 32 rows; 8 waves -> 256 rows.
    for (int ti = 0; ti < 2; ++ti) {
        const int tilebase = wv * 32 + ti * 16;
        const int r = row0 + tilebase + mrow;
        const float4* __restrict__ rowp = (const float4*)(sb + (size_t)r * D);

        v8f c = {};
        #pragma unroll 4
        for (int kt = 0; kt < D / 32; ++kt) {
            // A fragment (contiguous k order, same as B): this lane holds
            // K = kt*32 + khalf16 + {0..15}.
            const int k0 = kt * 32 + khalf16;
            float4 f0 = rowp[(k0 >> 2) + 0];
            float4 f1 = rowp[(k0 >> 2) + 1];
            float4 f2 = rowp[((k0 + 8) >> 2) + 0];
            float4 f3 = rowp[((k0 + 8) >> 2) + 1];
            v16bf a;
            a[0]=(__bf16)f0.x; a[1]=(__bf16)f0.y; a[2]=(__bf16)f0.z; a[3]=(__bf16)f0.w;
            a[4]=(__bf16)f1.x; a[5]=(__bf16)f1.y; a[6]=(__bf16)f1.z; a[7]=(__bf16)f1.w;
            a[8]=(__bf16)f2.x; a[9]=(__bf16)f2.y; a[10]=(__bf16)f2.z; a[11]=(__bf16)f2.w;
            a[12]=(__bf16)f3.x; a[13]=(__bf16)f3.y; a[14]=(__bf16)f3.z; a[15]=(__bf16)f3.w;

            // B fragment: every column of the 32x16 B holds the same hidden
            // slice, so each lane grabs 16 consecutive bf16 from LDS.
            v16bf bm = *(const v16bf*)&hidbf[kt * 32 + khalf16];

            c = __builtin_amdgcn_wmma_f32_16x16x32_bf16(
                    false, a, false, bm, (short)0, c, false, false);
        }
        // C layout: lane 0 (N=0) holds rows 0..7, lane 16 holds rows 8..15.
        if (lane == 0) {
            #pragma unroll
            for (int j = 0; j < 8; ++j) sc[tilebase + j] = c[j];
        }
        if (lane == 16) {
            #pragma unroll
            for (int j = 0; j < 8; ++j) sc[tilebase + 8 + j] = c[j];
        }
    }
    __syncthreads();

    // Local softmax stats over the 256 chunk scores (one score per thread).
    red[tid] = sc[tid];
    __syncthreads();
    for (int off = 128; off > 0; off >>= 1) {
        if (tid < off) red[tid] = fmaxf(red[tid], red[tid + off]);
        __syncthreads();
    }
    const float m = red[0];
    __syncthreads();
    const float w = __expf(sc[tid] - m);
    wbuf[tid] = w;
    red[tid]  = w;
    __syncthreads();
    for (int off = 128; off > 0; off >>= 1) {
        if (tid < off) red[tid] += red[tid + off];
        __syncthreads();
    }
    const float l = red[0];

    // Local weighted sum over the chunk: thread t owns d = 2t, 2t+1.
    // Second pass over the same 512KB chunk -> expected L2-resident.
    float2 acc = {0.f, 0.f};
    const float2* __restrict__ p2 = (const float2*)(sb + (size_t)row0 * D);
    #pragma unroll 4
    for (int s = 0; s < ROWS; ++s) {
        if (s + 8 < ROWS)
            __builtin_prefetch(p2 + (size_t)(s + 8) * (D / 2) + tid, 0, 0);
        const float ww = wbuf[s];
        const float2 v = p2[(size_t)s * (D / 2) + tid];
        acc.x = fmaf(ww, v.x, acc.x);
        acc.y = fmaf(ww, v.y, acc.y);
    }

    float* __restrict__ o = part + (size_t)blk * PART_STRIDE;
    if (tid == 0) { o[0] = m; o[1] = l; }
    o[2 + 2 * tid]     = acc.x;
    o[2 + 2 * tid + 1] = acc.y;
}

// ---------------------------------------------------------------------------
// Kernel 2: merge the CHUNKS partials per batch (flash rescale) and normalize.
// ---------------------------------------------------------------------------
__global__ __launch_bounds__(256)
void attn_reduce_kernel(const float* __restrict__ part,
                        float* __restrict__ out) {
    __shared__ float scale[CHUNKS];
    const int b   = blockIdx.x;
    const int tid = threadIdx.x;
    const float* __restrict__ pb = part + (size_t)b * CHUNKS * PART_STRIDE;

    if (tid == 0) {
        float M = -INFINITY;
        for (int c = 0; c < CHUNKS; ++c)
            M = fmaxf(M, pb[(size_t)c * PART_STRIDE]);
        float L = 0.f;
        for (int c = 0; c < CHUNKS; ++c)
            L += pb[(size_t)c * PART_STRIDE + 1] *
                 __expf(pb[(size_t)c * PART_STRIDE] - M);
        const float inv = 1.f / L;
        for (int c = 0; c < CHUNKS; ++c)
            scale[c] = __expf(pb[(size_t)c * PART_STRIDE] - M) * inv;
    }
    __syncthreads();

    float2 acc = {0.f, 0.f};
    for (int c = 0; c < CHUNKS; ++c) {
        const float2* __restrict__ a2 =
            (const float2*)(pb + (size_t)c * PART_STRIDE + 2);
        const float2 v = a2[tid];
        acc.x = fmaf(scale[c], v.x, acc.x);
        acc.y = fmaf(scale[c], v.y, acc.y);
    }
    out[(size_t)b * D + 2 * tid]     = acc.x;
    out[(size_t)b * D + 2 * tid + 1] = acc.y;
}

extern "C" void kernel_launch(void* const* d_in, const int* in_sizes, int n_in,
                              void* d_out, int out_size, void* d_ws, size_t ws_size,
                              hipStream_t stream) {
    const float* hidden = (const float*)d_in[0];  // [B, D]
    const float* src    = (const float*)d_in[1];  // [B, S, D]
    float* out  = (float*)d_out;                  // [B, D]
    float* part = (float*)d_ws;                   // [B*CHUNKS, D+2] ~2.1 MB

    attn_partial_kernel<<<B * CHUNKS, 256, 0, stream>>>(hidden, src, part);
    attn_reduce_kernel<<<B, 256, 0, stream>>>(part, out);
}
